// attn_mh_764504178975
// MI455X (gfx1250) — matrix operand order repaired and checked by reading, not compiled, not run on hardware
//
#include <hip/hip_runtime.h>
#include <hip/hip_bf16.h>

// ---------------------------------------------------------------------------
// MI455X (gfx1250, wave32) fused multi-head attention.
// GEMMs: V_WMMA_F32_16X16X32_F16.  f16 tile staging uses CDNA5
// GLOBAL_LOAD_ASYNC_TO_LDS_B128 (ASYNCcnt) with double-buffered pipelines;
// f32 sources are register-staged (prefetch overlaps XDL compute).
// ---------------------------------------------------------------------------

typedef __attribute__((ext_vector_type(16))) _Float16 v16h;
typedef __attribute__((ext_vector_type(8)))  _Float16 v8h;
typedef __attribute__((ext_vector_type(8)))  float    v8f;

#define BB   128
#define SS   512
#define HH   1000
#define NHH  8
#define DKK  125
#define DKP  128
#define MTOT (BB * SS)            // 65536
#define ZLD  1024                 // padded z row stride (f16)

// ---- CDNA5 async LDS DMA helpers ------------------------------------------

__device__ __forceinline__ unsigned lds_addr_of(const void* p) {
  return (unsigned)(unsigned long long)p;   // generic LDS addr: low 32 bits
}

// GVS mode: global_mem_addr = SADDR + VGPR_off + inst_offset; LDS dst in VDST.
__device__ __forceinline__ void async_b128(unsigned lds, const void* base,
                                           unsigned off) {
  asm volatile("global_load_async_to_lds_b128 %0, %1, %2 offset:0"
               :: "v"(lds), "v"(off), "s"(base)
               : "memory");
}

__device__ __forceinline__ void wait_async0() {
  asm volatile("s_wait_asynccnt 0x0" ::: "memory");
}

// ---- WMMA helpers ----------------------------------------------------------

__device__ __forceinline__ v8f wmma16(v16h a, v16h b, v8f c) {
  return __builtin_amdgcn_wmma_f32_16x16x32_f16(false, a, false, b, (short)0, c,
                                                false, false);
}

// A-matrix fragment (16x32 f16): lane L<16 -> M=L, K={0..7,16..23};
// lane L>=16 -> M=L-16, K={8..15,24..31}.  Source: row-major [rows][lda] LDS.
__device__ __forceinline__ v16h ld_frag_a(const _Float16* p0, int row0, int lda) {
  const int lane = threadIdx.x & 31;
  const _Float16* p = p0 + (row0 + (lane & 15)) * lda + ((lane >> 4) << 3);
  v16h r;
  ((v8h*)&r)[0] = ((const v8h*)p)[0];   // K  +0..7
  ((v8h*)&r)[1] = ((const v8h*)p)[2];   // K +16..23
  return r;
}

// B-matrix fragment (32x16 f16): lane L<16 -> N=L, K={0..7,16..23};
// lane L>=16 -> N=L-16, K={8..15,24..31}.  Source: N-major [N][lda] LDS.
__device__ __forceinline__ v16h ld_frag_b(const _Float16* p0, int row0, int lda) {
  const int lane = threadIdx.x & 31;
  const _Float16* p = p0 + (row0 + (lane & 15)) * lda + ((lane >> 4) << 3);
  v16h r;
  ((v8h*)&r)[0] = ((const v8h*)p)[0];
  ((v8h*)&r)[1] = ((const v8h*)p)[2];
  return r;
}

// ---- register-staged 128x32 tile: f32 global -> VGPRs -> f16 LDS ----------

__device__ __forceinline__ void load_regs_f32(float4 r[4], const float* src,
                                              int row0, int rmax, int k0, int K) {
  const int tid = threadIdx.x;
  if (k0 + 32 <= K && row0 + 128 <= rmax) {
#pragma unroll
    for (int i = 0; i < 4; ++i) {
      int chunk = tid + 256 * i;        // 1024 chunks of 4 floats
      int row = chunk >> 3, c4 = (chunk & 7) << 2;
      r[i] = *(const float4*)(src + (size_t)(row0 + row) * K + k0 + c4);
    }
  } else {
#pragma unroll
    for (int i = 0; i < 4; ++i) {
      int chunk = tid + 256 * i;
      int row = chunk >> 3, c4 = (chunk & 7) << 2;
      int gr = row0 + row, gc = k0 + c4;
      float4 v = make_float4(0.f, 0.f, 0.f, 0.f);
      if (gr < rmax) {
        const float* g = src + (size_t)gr * K;
        if (gc + 0 < K) v.x = g[gc + 0];
        if (gc + 1 < K) v.y = g[gc + 1];
        if (gc + 2 < K) v.z = g[gc + 2];
        if (gc + 3 < K) v.w = g[gc + 3];
      }
      r[i] = v;
    }
  }
}

__device__ __forceinline__ void store_regs_f16(_Float16* dst, const float4 r[4]) {
  const int tid = threadIdx.x;
#pragma unroll
  for (int i = 0; i < 4; ++i) {
    int chunk = tid + 256 * i;
    int row = chunk >> 3, c4 = (chunk & 7) << 2;
    _Float16* d = dst + row * 40 + c4;
    d[0] = (_Float16)r[i].x; d[1] = (_Float16)r[i].y;
    d[2] = (_Float16)r[i].z; d[3] = (_Float16)r[i].w;
  }
}

// ---- kernel 1: projection GEMM  out = act @ W^T + b ------------------------
// grid = (n-blocks, m-blocks): consecutive blocks share the A panel -> L2 reuse
// Register double-buffering: tile kt+1 is prefetched during the WMMAs of kt.
// mode 0: store f16 head-major  qh/kh[(bh*512+s)*128 + d]
// mode 1: store f16 transposed  vhT[(bh*128+d)*512 + s]
__global__ __launch_bounds__(256) void gemm_proj_kernel(
    const float* __restrict__ A, const float* __restrict__ W,
    const float* __restrict__ bias, _Float16* __restrict__ out, int mode) {
  const int n0 = blockIdx.x * 128;
  const int m0 = blockIdx.y * 128;
  const int tid = threadIdx.x;
  const int w = tid >> 5, lane = tid & 31;
  const int wm = w >> 1, wn = w & 1;          // 4x2 wave grid -> 32x64 per wave
  const int hi = lane >> 4, ln = lane & 15;

  __shared__ _Float16 As[128 * 40];
  __shared__ _Float16 Bs[128 * 40];

  v8f acc[2][4] = {};
  float4 ra[4], rb[4];
  load_regs_f32(ra, A, m0, MTOT, 0, HH);
  load_regs_f32(rb, W, n0, HH, 0, HH);

  for (int kt = 0; kt < 32; ++kt) {           // K = 1000, padded to 1024
    store_regs_f16(As, ra);
    store_regs_f16(Bs, rb);
    __syncthreads();
    if (kt + 1 < 32) {                        // prefetch next tile into VGPRs
      load_regs_f32(ra, A, m0, MTOT, (kt + 1) * 32, HH);
      load_regs_f32(rb, W, n0, HH, (kt + 1) * 32, HH);
    }
    v16h a0 = ld_frag_a(As, wm * 32 + 0, 40);
    v16h a1 = ld_frag_a(As, wm * 32 + 16, 40);
#pragma unroll
    for (int nt = 0; nt < 4; ++nt) {
      v16h b = ld_frag_b(Bs, wn * 64 + nt * 16, 40);
      acc[0][nt] = wmma16(a0, b, acc[0][nt]);
      acc[1][nt] = wmma16(a1, b, acc[1][nt]);
    }
    __syncthreads();
  }

#pragma unroll
  for (int mt = 0; mt < 2; ++mt)
#pragma unroll
    for (int nt = 0; nt < 4; ++nt) {
      const int cn = n0 + wn * 64 + nt * 16 + ln;
      if (cn >= HH) continue;
      const int h = cn / DKK, d = cn - h * DKK;
      const float bv = bias[cn];
#pragma unroll
      for (int r = 0; r < 8; ++r) {
        const int cm = m0 + wm * 32 + mt * 16 + hi * 8 + r;
        const int b = cm >> 9, s = cm & 511;
        const float val = acc[mt][nt][r] + bv;
        const size_t off =
            (mode == 0) ? ((size_t)((b * NHH + h) * SS + s) * DKP + d)
                        : ((size_t)((b * NHH + h) * DKP + d) * SS + s);
        out[off] = (_Float16)val;
      }
    }
}

// ---- kernel 2: attention (scores + mask + softmax + P@V) -------------------
// One block: (b,h) head, 32 queries.  K and V phases share one pair of LDS
// buffers and are both double-buffered async-DMA pipelines (1 barrier/iter).
#define ATTN_SMEM_BYTES 145536

__global__ __launch_bounds__(256) void attn_kernel(
    const _Float16* __restrict__ qh, const _Float16* __restrict__ kh,
    const _Float16* __restrict__ vhT, const int* __restrict__ mask,
    _Float16* __restrict__ zout) {
  const int blk = blockIdx.x;
  const int qb = blk & 15, bh = blk >> 4;
  const int b = bh >> 3, h = bh & 7;
  const int q0 = qb * 32;
  const int tid = threadIdx.x, w = tid >> 5, lane = tid & 31;
  const int wm = w & 1, wn = w >> 1;          // 2x4 wave grid
  const int hi = lane >> 4, ln = lane & 15;

  extern __shared__ char smem_raw[];
  _Float16* Qs  = (_Float16*)smem_raw;        // 32*136 = 4352 h
  _Float16* KV0 = Qs + 32 * 136;              // 8704 h (K: 64x136 / V: 128x40)
  _Float16* KV1 = KV0 + 64 * 136;             // 8704 h
  _Float16* Ps  = KV1 + 64 * 136;             // 32*520 (probs, f16)
  float* Ss     = (float*)(Ps + 32 * 520);    // 32*520 (scores, f32)
  float* redmax = Ss + 32 * 520;              // 256
  float* redsum = redmax + 256;               // 256
  float* rowinv = redsum + 256;               // 32

  auto issue_k = [&](int kb, _Float16* dstb) {
    const unsigned base_k =
        (unsigned)((((unsigned)bh * SS + kb * 64) * DKP) * 2);
#pragma unroll
    for (int i = 0; i < 4; ++i) {
      int chunk = tid + 256 * i;              // 1024 chunks of 16B
      int row = chunk >> 4, c8 = (chunk & 15) << 3;
      async_b128(lds_addr_of(dstb + row * 136 + c8), kh,
                 base_k + (unsigned)((row * DKP + c8) * 2));
    }
  };
  auto issue_v = [&](int kt, _Float16* dstb) {
    const unsigned base_v = (unsigned)(((unsigned)bh * DKP * SS) * 2);
#pragma unroll
    for (int i = 0; i < 2; ++i) {
      int chunk = tid + 256 * i;              // 512 chunks: 128 rows x 32 k
      int row = chunk >> 2, c8 = (chunk & 3) << 3;
      async_b128(lds_addr_of(dstb + row * 40 + c8), vhT,
                 base_v + (unsigned)((row * SS + kt * 32 + c8) * 2));
    }
  };

  // prologue: DMA Q tile (32x128) + first K tile
  {
    const unsigned base_q = (unsigned)((((unsigned)bh * SS + q0) * DKP) * 2);
#pragma unroll
    for (int i = 0; i < 2; ++i) {
      int chunk = tid + 256 * i;              // 512 chunks of 16B
      int row = chunk >> 4, c8 = (chunk & 15) << 3;
      async_b128(lds_addr_of(Qs + row * 136 + c8), qh,
                 base_q + (unsigned)((row * DKP + c8) * 2));
    }
  }
  issue_k(0, KV0);
  wait_async0();
  __syncthreads();

  // hoist the 4 Q A-fragments into registers (reused for all 8 key blocks)
  v16h aq[4];
#pragma unroll
  for (int kt = 0; kt < 4; ++kt) aq[kt] = ld_frag_a(Qs + kt * 32, wm * 16, 136);

  // scores: double-buffered async K pipeline
  for (int kb = 0; kb < 8; ++kb) {
    if (kb + 1 < 8) issue_k(kb + 1, (kb & 1) ? KV0 : KV1);
    const _Float16* Kcur = (kb & 1) ? KV1 : KV0;
    v8f acc = {};
#pragma unroll
    for (int kt = 0; kt < 4; ++kt)
      acc = wmma16(aq[kt], ld_frag_b(Kcur + kt * 32, wn * 16, 136), acc);
    const float scale = 0.08944271909999159f; // 1/sqrt(125)
    const int col = kb * 64 + wn * 16 + ln;
#pragma unroll
    for (int r = 0; r < 8; ++r)
      Ss[(wm * 16 + hi * 8 + r) * 520 + col] = acc[r] * scale;
    wait_async0();                            // next K tile (own loads) done
    __syncthreads();                          // publish tile + Ss writes
  }

  // stream first V tile under the softmax ALU work
  issue_v(0, KV0);

  // masked softmax: 8 threads per query row, 64 cols each
  {
    const int row = tid >> 3, seg = tid & 7;
    const int* mrow = mask + (size_t)b * SS;
    float lmax = -3.0e38f;
    for (int j = 0; j < 64; ++j) {
      const int col = seg * 64 + j;
      float v = Ss[row * 520 + col];
      if (mrow[col] != 0) v = -10000.0f;
      Ss[row * 520 + col] = v;
      lmax = fmaxf(lmax, v);
    }
    redmax[tid] = lmax;
    __syncthreads();
    float rmax = -3.0e38f;
    for (int s2 = 0; s2 < 8; ++s2) rmax = fmaxf(rmax, redmax[(row << 3) + s2]);
    float lsum = 0.f;
    for (int j = 0; j < 64; ++j) {
      const int col = seg * 64 + j;
      const float e = __expf(Ss[row * 520 + col] - rmax);
      Ps[row * 520 + col] = (_Float16)e;     // unnormalized; 1/sum in epilogue
      lsum += e;
    }
    redsum[tid] = lsum;
    __syncthreads();
    if (seg == 0) {
      float s = 0.f;
      for (int s2 = 0; s2 < 8; ++s2) s += redsum[(row << 3) + s2];
      rowinv[row] = 1.0f / s;
    }
  }

  // z = P @ V (vhT layout): double-buffered async pipeline over 16 k-steps
  v8f zacc[2] = {};
  for (int kt = 0; kt < 16; ++kt) {
    wait_async0();                            // V tile kt (own loads) done
    __syncthreads();                          // all waves ready; prev compute done
    if (kt + 1 < 16) issue_v(kt + 1, ((kt + 1) & 1) ? KV1 : KV0);
    const _Float16* Vcur = (kt & 1) ? KV1 : KV0;
    v16h a = ld_frag_a(Ps + kt * 32, wm * 16, 520);
#pragma unroll
    for (int nt = 0; nt < 2; ++nt) {
      v16h bv = ld_frag_b(Vcur, (wn * 2 + nt) * 16, 40);
      zacc[nt] = wmma16(a, bv, zacc[nt]);
    }
  }

#pragma unroll
  for (int nt = 0; nt < 2; ++nt) {
    const int d = wn * 32 + nt * 16 + ln;
    if (d >= DKK) continue;
#pragma unroll
    for (int r = 0; r < 8; ++r) {
      const int mloc = wm * 16 + hi * 8 + r;
      const float v = zacc[nt][r] * rowinv[mloc];
      zout[((size_t)b * SS + q0 + mloc) * ZLD + h * DKK + d] = (_Float16)v;
    }
  }
}

// ---- kernel 3: output projection  y = z @ Wo^T + b  (f32 into d_out) -------
// A tiles: double-buffered async LDS DMA.  W tiles: register-staged f32->f16.
__global__ __launch_bounds__(256) void gemm_out_kernel(
    const _Float16* __restrict__ Z, const float* __restrict__ W,
    const float* __restrict__ bias, float* __restrict__ out) {
  const int n0 = blockIdx.x * 128;
  const int m0 = blockIdx.y * 128;
  const int tid = threadIdx.x;
  const int w = tid >> 5, lane = tid & 31;
  const int wm = w >> 1, wn = w & 1;
  const int hi = lane >> 4, ln = lane & 15;

  __shared__ _Float16 As2[2][128 * 40];
  __shared__ _Float16 Bs[128 * 40];

  auto issue_a = [&](int kt, int buf) {
    const int k0 = kt * 32;
#pragma unroll
    for (int i = 0; i < 2; ++i) {
      int chunk = tid + 256 * i;              // 512 chunks of 16B
      int row = chunk >> 2, c8 = (chunk & 3) << 3;
      async_b128(lds_addr_of(&As2[buf][row * 40 + c8]), Z,
                 (unsigned)((((m0 + row) * ZLD) + k0 + c8) * 2));
    }
  };

  v8f acc[2][4] = {};
  float4 rb[4];
  issue_a(0, 0);
  load_regs_f32(rb, W, n0, HH, 0, HH);

  for (int kt = 0; kt < 32; ++kt) {
    store_regs_f16(Bs, rb);
    wait_async0();                            // A tile kt (own loads) done
    __syncthreads();                          // As2[kt&1] + Bs visible
    if (kt + 1 < 32) {                        // prefetch next step
      issue_a(kt + 1, (kt + 1) & 1);
      load_regs_f32(rb, W, n0, HH, (kt + 1) * 32, HH);
    }
    v16h a0 = ld_frag_a(As2[kt & 1], wm * 32 + 0, 40);
    v16h a1 = ld_frag_a(As2[kt & 1], wm * 32 + 16, 40);
#pragma unroll
    for (int nt = 0; nt < 4; ++nt) {
      v16h b = ld_frag_b(Bs, wn * 64 + nt * 16, 40);
      acc[0][nt] = wmma16(a0, b, acc[0][nt]);
      acc[1][nt] = wmma16(a1, b, acc[1][nt]);
    }
    __syncthreads();                          // done with Bs before overwrite
  }

#pragma unroll
  for (int mt = 0; mt < 2; ++mt)
#pragma unroll
    for (int nt = 0; nt < 4; ++nt) {
      const int cn = n0 + wn * 64 + nt * 16 + ln;
      if (cn >= HH) continue;
      const float bv = bias[cn];
#pragma unroll
      for (int r = 0; r < 8; ++r) {
        const int cm = m0 + wm * 32 + mt * 16 + hi * 8 + r;
        out[(size_t)cm * HH + cn] = acc[mt][nt][r] + bv;
      }
    }
}

// ---- kernel 4: residual + layernorm (in-place on d_out) --------------------
__global__ __launch_bounds__(256) void ln_kernel(const float* __restrict__ x,
                                                 const float* __restrict__ g,
                                                 const float* __restrict__ bt,
                                                 float* __restrict__ y) {
  const int row = blockIdx.x;
  const int tid = threadIdx.x;
  __shared__ float rs[256], rq[256];
  const float* xr = x + (size_t)row * HH;
  float* yr = y + (size_t)row * HH;
  float s = 0.f, sq = 0.f;
  for (int i = tid; i < HH; i += 256) {
    const float v = xr[i] + yr[i];
    s += v; sq += v * v;
  }
  rs[tid] = s; rq[tid] = sq;
  __syncthreads();
  for (int off = 128; off > 0; off >>= 1) {
    if (tid < off) { rs[tid] += rs[tid + off]; rq[tid] += rq[tid + off]; }
    __syncthreads();
  }
  const float mu = rs[0] * (1.0f / HH);
  const float var = rq[0] * (1.0f / HH) - mu * mu;
  const float rstd = rsqrtf(var + 1e-5f);
  for (int i = tid; i < HH; i += 256) {
    const float v = xr[i] + yr[i];
    yr[i] = (v - mu) * rstd * g[i] + bt[i];
  }
}

// ---- host launch -----------------------------------------------------------

extern "C" void kernel_launch(void* const* d_in, const int* in_sizes, int n_in,
                              void* d_out, int out_size, void* d_ws, size_t ws_size,
                              hipStream_t stream) {
  const float* q    = (const float*)d_in[0];
  const float* k    = (const float*)d_in[1];
  const float* v    = (const float*)d_in[2];
  const int*   mask = (const int*)d_in[3];
  const float* Wq_w = (const float*)d_in[4];
  const float* Wq_b = (const float*)d_in[5];
  const float* Wk_w = (const float*)d_in[6];
  const float* Wk_b = (const float*)d_in[7];
  const float* Wv_w = (const float*)d_in[8];
  const float* Wv_b = (const float*)d_in[9];
  const float* Wo_w = (const float*)d_in[10];
  const float* Wo_b = (const float*)d_in[11];
  const float* ln_g = (const float*)d_in[12];
  const float* ln_b = (const float*)d_in[13];
  (void)in_sizes; (void)n_in; (void)out_size; (void)ws_size;

  const size_t HEAD_BYTES = (size_t)BB * NHH * SS * DKP * 2; // 128 MiB each
  const size_t Z_BYTES    = (size_t)MTOT * ZLD * 2;          // 128 MiB
  char* ws = (char*)d_ws;
  _Float16* qh  = (_Float16*)(ws + 0 * HEAD_BYTES);
  _Float16* kh  = (_Float16*)(ws + 1 * HEAD_BYTES);
  _Float16* vhT = (_Float16*)(ws + 2 * HEAD_BYTES);
  _Float16* z   = (_Float16*)(ws + 3 * HEAD_BYTES);   // [65536][1024] f16

  // zero qh/kh/vhT (DK 125->128 pad) and z (col 1000->1024 pad)
  hipMemsetAsync(ws, 0, 3 * HEAD_BYTES + Z_BYTES, stream);

  const dim3 ggrid((HH + 127) / 128, MTOT / 128);     // (8, 512): n fastest
  gemm_proj_kernel<<<ggrid, 256, 0, stream>>>(q, Wq_w, Wq_b, qh, 0);
  gemm_proj_kernel<<<ggrid, 256, 0, stream>>>(k, Wk_w, Wk_b, kh, 0);
  gemm_proj_kernel<<<ggrid, 256, 0, stream>>>(v, Wv_w, Wv_b, vhT, 1);

  hipFuncSetAttribute(reinterpret_cast<const void*>(attn_kernel),
                      hipFuncAttributeMaxDynamicSharedMemorySize,
                      ATTN_SMEM_BYTES);
  attn_kernel<<<BB * NHH * (SS / 32), 256, ATTN_SMEM_BYTES, stream>>>(
      qh, kh, vhT, mask, z);

  gemm_out_kernel<<<ggrid, 256, 0, stream>>>(z, Wo_w, Wo_b, (float*)d_out);
  ln_kernel<<<MTOT, 256, 0, stream>>>(q, ln_g, ln_b, (float*)d_out);
}
